// GQA_37718402794138
// MI455X (gfx1250) — compile-verified, matrix operand order checked by reading, not run on hardware
//
#include <hip/hip_runtime.h>
#include <hip/hip_bf16.h>

typedef __attribute__((ext_vector_type(16))) __bf16 v16bf;
typedef __attribute__((ext_vector_type(8)))  __bf16 v8bf;
typedef __attribute__((ext_vector_type(8)))  float  v8f;

#define HID 2048
#define SEQ 2048
#define BATCH 2
#define NH 16
#define NG 4
#define HD 128
#define KVD (NG * HD) /* 512 */

// ---------------------------------------------------------------------------
// CDNA5 async copy: global -> LDS, 16 bytes per lane, tracked by ASYNCcnt.
// LDS destination address = low 32 bits of the generic shared pointer
// (flat LDS addressing truncates to addr[31:0]).
// ---------------------------------------------------------------------------
__device__ __forceinline__ void async_b128(void* lds, const void* g) {
  uint32_t laddr = (uint32_t)(uintptr_t)lds;
  uint64_t gaddr = (uint64_t)(uintptr_t)g;
  asm volatile("global_load_async_to_lds_b128 %0, %1, off"
               :: "v"(laddr), "v"(gaddr) : "memory");
}
__device__ __forceinline__ void wait_async0() {
  asm volatile("s_wait_asynccnt 0x0" ::: "memory");
}

// ---------------------------------------------------------------------------
// Fragment loader for K-contiguous (row-major over K) matrices (global or LDS).
// CDNA5 16-bit A-matrix 16x32 layout: lane l (0..15) holds row l with
// K = {0..7, 16..23}; lane l+16 holds row l with K = {8..15, 24..31}.
// B-fragment (32x16) mirrors it with "row" = B-column.
// ---------------------------------------------------------------------------
__device__ __forceinline__ v16bf ld_frag(const __bf16* __restrict__ base,
                                         int ldk, int l, int hi) {
  const __bf16* p = base + (size_t)l * ldk + hi * 8;
  v8bf a = *(const v8bf*)p;
  v8bf b = *(const v8bf*)(p + 16);
  return __builtin_shufflevector(a, b, 0, 1, 2, 3, 4, 5, 6, 7,
                                 8, 9, 10, 11, 12, 13, 14, 15);
}

__device__ __forceinline__ v8f wmma_bf16(v16bf a, v16bf b, v8f c) {
  return __builtin_amdgcn_wmma_f32_16x16x32_bf16(false, a, false, b,
                                                 (short)0, c, false, false);
}

// ---------------------------------------------------------------------------
// f32 -> bf16 conversion (grid-stride)
// ---------------------------------------------------------------------------
__global__ void cvt_kernel(const float* __restrict__ in,
                           __bf16* __restrict__ out, int n) {
  int i = blockIdx.x * blockDim.x + threadIdx.x;
  int stride = gridDim.x * blockDim.x;
  for (; i < n; i += stride) out[i] = (__bf16)in[i];
}

// ---------------------------------------------------------------------------
// C[M,N] = A[M,K] @ W[N,K]^T + bias.
// Block = 8 waves, 64(M) x 128(N) tile; per-wave 32x32 (2x2 WMMA tiles).
// A/W K-step tiles staged to LDS with async copies, double buffered.
// ---------------------------------------------------------------------------
__global__ __launch_bounds__(256, 1) void gemm_bf16_kernel(
    const __bf16* __restrict__ A, const __bf16* __restrict__ W,
    const float* __restrict__ bias, __bf16* __restrict__ outb,
    float* __restrict__ outf, int M, int N, int K, int store_f32) {
  __shared__ __bf16 At[2][64][32];    // 8 KB
  __shared__ __bf16 Wt[2][128][32];   // 16 KB

  int tid = threadIdx.x;
  int lane = tid & 31, wid = tid >> 5;  // wave32
  int l = lane & 15, hi = lane >> 4;
  int m0 = blockIdx.y * 64, n0 = blockIdx.x * 128;
  int wm = (wid >> 2) * 32, wn = (wid & 3) * 32;  // wave tile inside block

  // staging assignment: A 64x32 = 256 x b128 (1/thread), W 128x32 = 512 (2/thread)
  int aRow = tid >> 2, aChk = (tid & 3) * 8;

  v8f acc00 = {}, acc01 = {}, acc10 = {}, acc11 = {};
  const __bf16* a0 = A + (size_t)m0 * K;
  const __bf16* w0 = W + (size_t)n0 * K;

  auto stage = [&](int buf, int kb) {
    async_b128(&At[buf][aRow][aChk], a0 + (size_t)aRow * K + kb + aChk);
#pragma unroll
    for (int j = 0; j < 2; ++j) {
      int idx = tid + 256 * j;
      int wRow = idx >> 2, wChk = (idx & 3) * 8;
      async_b128(&Wt[buf][wRow][wChk], w0 + (size_t)wRow * K + kb + wChk);
    }
  };

  stage(0, 0);
  wait_async0();
  __syncthreads();

  int buf = 0;
  for (int kb = 0; kb < K; kb += 32) {
    if (kb + 32 < K) stage(buf ^ 1, kb + 32);  // overlap with compute below

    v16bf af0 = ld_frag(&At[buf][wm][0], 32, l, hi);
    v16bf af1 = ld_frag(&At[buf][wm + 16][0], 32, l, hi);
    v16bf bf0 = ld_frag(&Wt[buf][wn][0], 32, l, hi);
    v16bf bf1 = ld_frag(&Wt[buf][wn + 16][0], 32, l, hi);
    acc00 = wmma_bf16(af0, bf0, acc00);
    acc01 = wmma_bf16(af0, bf1, acc01);
    acc10 = wmma_bf16(af1, bf0, acc10);
    acc11 = wmma_bf16(af1, bf1, acc11);

    wait_async0();     // next buffer staged
    __syncthreads();   // everyone done reading current buffer + staging visible
    buf ^= 1;
  }

  // C tile layout: element v of lane (l,hi) -> row 16i+v+hi*8, col 16j+l
  for (int j = 0; j < 2; ++j) {
    float bval = bias ? bias[n0 + wn + 16 * j + l] : 0.0f;
    for (int i = 0; i < 2; ++i) {
      v8f acc = (i == 0) ? (j == 0 ? acc00 : acc01) : (j == 0 ? acc10 : acc11);
#pragma unroll
      for (int v = 0; v < 8; ++v) {
        size_t row = (size_t)(m0 + wm + 16 * i + v + hi * 8);
        size_t col = (size_t)(n0 + wn + 16 * j + l);
        float val = acc[v] + bval;
        if (store_f32)
          outf[row * N + col] = val;
        else
          outb[row * N + col] = (__bf16)val;
      }
    }
  }
}

// ---------------------------------------------------------------------------
// V transpose: Vh[(b*S+s), g*HD+d] -> Vt[((b*G+g)*HD+d), s]
// ---------------------------------------------------------------------------
__global__ void vtrans_kernel(const __bf16* __restrict__ Vh,
                              __bf16* __restrict__ Vt) {
  int idx = blockIdx.x * blockDim.x + threadIdx.x;  // B*S*KVD = 2097152
  int d = idx & 127;
  int g = (idx >> 7) & 3;
  int s = (idx >> 9) & 2047;
  int bb = idx >> 20;
  Vt[((size_t)(bb * NG + g) * HD + d) * SEQ + s] = Vh[idx];
}

// ---------------------------------------------------------------------------
// Flash attention. grid = (S/128, H, B), block = 256 (8 waves); each wave
// owns a 16-row q strip. Per 32-key step the block stages the SHARED K chunk
// (32x128) and Vt chunk (128x32) to LDS with async copies (8x reuse across
// waves), double buffered; online softmax in f32; P goes through a per-wave
// LDS tile to convert C-layout -> A-fragment layout.
// ---------------------------------------------------------------------------
__global__ __launch_bounds__(256, 1) void flash_kernel(
    const __bf16* __restrict__ Q, const __bf16* __restrict__ Km,
    const __bf16* __restrict__ Vt, __bf16* __restrict__ Mid) {
  __shared__ __bf16 Kt[2][32][128];   // 16 KB: keys x d
  __shared__ __bf16 Vtt[2][128][32];  // 16 KB: d x keys
  __shared__ __bf16 Pl[8][16][32];    // 8 KB per-wave P tiles

  int tid = threadIdx.x;
  int lane = tid & 31, wid = tid >> 5;
  int l = lane & 15, hi = lane >> 4;
  int h = blockIdx.y, bb = blockIdx.z;
  int g = h >> 2;  // rep = H/G = 4
  int q0 = blockIdx.x * 128 + wid * 16;
  const float sc = 0.08838834764831845f;  // 1/sqrt(128)

  const __bf16* qbase = Q + (size_t)(bb * SEQ + q0) * HID + h * HD;
  const __bf16* kbase = Km + (size_t)bb * SEQ * KVD + g * HD;
  const __bf16* vtb = Vt + (size_t)(bb * NG + g) * HD * SEQ;

  v16bf qf[4];
#pragma unroll
  for (int c = 0; c < 4; ++c) qf[c] = ld_frag(qbase + c * 32, HID, l, hi);

  v8f o[8] = {};
  float mI[8], lI[8];
#pragma unroll
  for (int v = 0; v < 8; ++v) { mI[v] = -3.0e38f; lI[v] = 0.0f; }

  // staging: K 32x128 = 512 x b128, Vt 128x32 = 512 x b128 -> 4/thread
  auto stage = [&](int buf, int k0) {
#pragma unroll
    for (int j = 0; j < 2; ++j) {
      int idx = tid + 256 * j;
      int kRow = idx >> 4, kChk = (idx & 15) * 8;
      async_b128(&Kt[buf][kRow][kChk],
                 kbase + (size_t)(k0 + kRow) * KVD + kChk);
      int vRow = idx >> 2, vChk = (idx & 3) * 8;
      async_b128(&Vtt[buf][vRow][vChk],
                 vtb + (size_t)vRow * SEQ + k0 + vChk);
    }
  };

  stage(0, 0);
  wait_async0();
  __syncthreads();

  int buf = 0;
  for (int k0 = 0; k0 < SEQ; k0 += 32) {
    if (k0 + 32 < SEQ) stage(buf ^ 1, k0 + 32);

    // ---- scores: 2 tiles of 16x16 from LDS-staged K ----
    v8f s0 = {}, s1 = {};
#pragma unroll
    for (int c = 0; c < 4; ++c) {
      v16bf kf0 = ld_frag(&Kt[buf][0][c * 32], 128, l, hi);
      v16bf kf1 = ld_frag(&Kt[buf][16][c * 32], 128, l, hi);
      s0 = wmma_bf16(qf[c], kf0, s0);
      s1 = wmma_bf16(qf[c], kf1, s1);
    }

    // ---- online softmax; row r = v + hi*8 lives across lanes l=0..15 ----
    float f[8], p0[8], p1[8];
#pragma unroll
    for (int v = 0; v < 8; ++v) {
      float a = s0[v] * sc, b = s1[v] * sc;
      float cm = fmaxf(a, b);
      for (int m = 1; m < 16; m <<= 1) cm = fmaxf(cm, __shfl_xor(cm, m, 32));
      float mn = fmaxf(mI[v], cm);
      f[v] = __expf(mI[v] - mn);
      mI[v] = mn;
      p0[v] = __expf(a - mn);
      p1[v] = __expf(b - mn);
      float ps = p0[v] + p1[v];
      for (int m = 1; m < 16; m <<= 1) ps += __shfl_xor(ps, m, 32);
      lI[v] = lI[v] * f[v] + ps;
    }
#pragma unroll
    for (int dc = 0; dc < 8; ++dc)
#pragma unroll
      for (int v = 0; v < 8; ++v) o[dc][v] *= f[v];

    // ---- P: C-layout -> LDS -> A-fragment layout (DS in-order per wave) ----
#pragma unroll
    for (int v = 0; v < 8; ++v) {
      Pl[wid][v + hi * 8][l] = (__bf16)p0[v];
      Pl[wid][v + hi * 8][16 + l] = (__bf16)p1[v];
    }
    v8bf pa = *(const v8bf*)&Pl[wid][l][hi * 8];
    v8bf pb = *(const v8bf*)&Pl[wid][l][16 + hi * 8];
    v16bf pf = __builtin_shufflevector(pa, pb, 0, 1, 2, 3, 4, 5, 6, 7,
                                       8, 9, 10, 11, 12, 13, 14, 15);

    // ---- O += P(16x32) @ V(32x128) from LDS-staged Vt ----
#pragma unroll
    for (int dc = 0; dc < 8; ++dc) {
      v16bf vf = ld_frag(&Vtt[buf][dc * 16][0], 32, l, hi);
      o[dc] = wmma_bf16(pf, vf, o[dc]);
    }

    wait_async0();
    __syncthreads();
    buf ^= 1;
  }

  float inv[8];
#pragma unroll
  for (int v = 0; v < 8; ++v) inv[v] = 1.0f / lI[v];
  __bf16* mout = Mid + (size_t)(bb * SEQ + q0) * HID + h * HD;
#pragma unroll
  for (int dc = 0; dc < 8; ++dc)
#pragma unroll
    for (int v = 0; v < 8; ++v)
      mout[(size_t)(v + hi * 8) * HID + dc * 16 + l] = (__bf16)(o[dc][v] * inv[v]);
}

// ---------------------------------------------------------------------------
extern "C" void kernel_launch(void* const* d_in, const int* in_sizes, int n_in,
                              void* d_out, int out_size, void* d_ws,
                              size_t ws_size, hipStream_t stream) {
  (void)in_sizes; (void)n_in; (void)out_size; (void)ws_size;
  const float* X  = (const float*)d_in[0];
  const float* Wq = (const float*)d_in[1];
  const float* bq = (const float*)d_in[2];
  const float* Wk = (const float*)d_in[3];
  const float* bk = (const float*)d_in[4];
  const float* Wv = (const float*)d_in[5];
  const float* bv = (const float*)d_in[6];
  const float* Wo = (const float*)d_in[7];
  const float* bo = (const float*)d_in[8];

  const size_t MT = (size_t)BATCH * SEQ;  // 4096 rows
  char* p = (char*)d_ws;
  auto take = [&](size_t elems) {
    __bf16* r = (__bf16*)p;
    p += elems * sizeof(__bf16);
    return r;
  };
  __bf16* Xh  = take(MT * HID);
  __bf16* Wqh = take((size_t)HID * HID);
  __bf16* Wkh = take((size_t)KVD * HID);
  __bf16* Wvh = take((size_t)KVD * HID);
  __bf16* Woh = take((size_t)HID * HID);
  __bf16* Qh  = take(MT * HID);
  __bf16* Kh  = take(MT * KVD);
  __bf16* Vh  = take(MT * KVD);
  __bf16* Vt  = take(MT * KVD);
  __bf16* Mid = take(MT * HID);

  cvt_kernel<<<4096, 256, 0, stream>>>(X, Xh, (int)(MT * HID));
  cvt_kernel<<<4096, 256, 0, stream>>>(Wq, Wqh, HID * HID);
  cvt_kernel<<<1024, 256, 0, stream>>>(Wk, Wkh, KVD * HID);
  cvt_kernel<<<1024, 256, 0, stream>>>(Wv, Wvh, KVD * HID);
  cvt_kernel<<<4096, 256, 0, stream>>>(Wo, Woh, HID * HID);

  gemm_bf16_kernel<<<dim3(HID / 128, MT / 64), 256, 0, stream>>>(
      Xh, Wqh, bq, Qh, nullptr, (int)MT, HID, HID, 0);
  gemm_bf16_kernel<<<dim3(KVD / 128, MT / 64), 256, 0, stream>>>(
      Xh, Wkh, bk, Kh, nullptr, (int)MT, KVD, HID, 0);
  gemm_bf16_kernel<<<dim3(KVD / 128, MT / 64), 256, 0, stream>>>(
      Xh, Wvh, bv, Vh, nullptr, (int)MT, KVD, HID, 0);

  vtrans_kernel<<<(int)((MT * KVD) / 256), 256, 0, stream>>>(Vh, Vt);

  flash_kernel<<<dim3(SEQ / 128, NH, BATCH), 256, 0, stream>>>(Qh, Kh, Vt, Mid);

  gemm_bf16_kernel<<<dim3(HID / 128, MT / 64), 256, 0, stream>>>(
      Mid, Woh, bo, nullptr, (float*)d_out, (int)MT, HID, HID, 1);
}
